// Hebbian_10213432230645
// MI455X (gfx1250) — matrix operand order repaired and checked by reading, not compiled, not run on hardware
//
#include <hip/hip_runtime.h>

// ---------------------------------------------------------------------------
// Hebbian fast-weight update for MI455X (gfx1250, wave32, WMMA bf16 16x16x32)
// Pipeline: fuse1 -> fuse2 -> act -> stats -> hebb
// ---------------------------------------------------------------------------

#define BSZ   4096
#define TT    32
#define KD    128
#define VD    128
#define NROWS (BSZ * TT)        // 131072
#define LRATE 0.5f

typedef __bf16 v16bf __attribute__((ext_vector_type(16)));
typedef float  v8f   __attribute__((ext_vector_type(8)));
typedef unsigned int v4u __attribute__((ext_vector_type(4)));
typedef int v4i_vs __attribute__((vector_size(4 * sizeof(int))));

union FragBF { v16bf v; v4u q[2]; };

// ---- gfx1250 async global->LDS path (guarded: falls back to plain copies) --
#if defined(__gfx1250__) && __has_builtin(__builtin_amdgcn_global_load_async_to_lds_b128)
#define HAVE_ASYNC_LDS 1
#else
#define HAVE_ASYNC_LDS 0
#endif

#define ASG __attribute__((address_space(1)))
#define ASL __attribute__((address_space(3)))

__device__ __forceinline__ void async_b128(void* lds_dst, const void* gsrc) {
#if HAVE_ASYNC_LDS
  __builtin_amdgcn_global_load_async_to_lds_b128(
      (ASG v4i_vs*)gsrc, (ASL v4i_vs*)lds_dst, 0, 0);
#else
  *(v4u*)lds_dst = *(const v4u*)gsrc;
#endif
}

__device__ __forceinline__ void async_wait0() {
#if HAVE_ASYNC_LDS
#if __has_builtin(__builtin_amdgcn_s_wait_asynccnt)
  __builtin_amdgcn_s_wait_asynccnt(0);
#else
  asm volatile("s_wait_asynccnt 0x0" ::: "memory");
#endif
#endif
}

__device__ __forceinline__ unsigned short f2bf(float f) {
  unsigned int u = __float_as_uint(f);
  u += 0x7FFFu + ((u >> 16) & 1u);          // round-to-nearest-even
  return (unsigned short)(u >> 16);
}
__device__ __forceinline__ float bf2f(unsigned short h) {
  return __uint_as_float(((unsigned int)h) << 16);
}

// A-matrix fragment (16x32 bf16), src row-major [M][Kdim], per-lane: two 16B runs
__device__ __forceinline__ v16bf load_a_frag(const unsigned short* p, int stride,
                                             int m0, int kk, int lane) {
  int m = lane & 15, g = lane >> 4;
  const unsigned short* base = p + (m0 + m) * stride + kk + g * 8;
  FragBF f;
  f.q[0] = *(const v4u*)(base);
  f.q[1] = *(const v4u*)(base + 16);
  return f.v;
}

// B-matrix fragment (32x16 bf16) from the TRANSPOSED operand QT row-major [N][Kdim]
__device__ __forceinline__ v16bf load_b_frag(const unsigned short* p, int stride,
                                             int n0, int kk, int lane) {
  int n = lane & 15, g = lane >> 4;
  const unsigned short* base = p + (n0 + n) * stride + kk + g * 8;
  FragBF f;
  f.q[0] = *(const v4u*)(base);
  f.q[1] = *(const v4u*)(base + 16);
  return f.v;
}

__device__ __forceinline__ v8f wmma_bf16(v16bf a, v16bf b, v8f c) {
  return __builtin_amdgcn_wmma_f32_16x16x32_bf16(false, a, false, b,
                                                 (short)0, c, false, false);
}

__device__ __forceinline__ void store_tile_bf(unsigned short* dst, int stride,
                                              int m0, int n0, v8f d, int lane) {
  int g = lane >> 4, n = lane & 15;
#pragma unroll
  for (int r = 0; r < 8; ++r)
    dst[(m0 + g * 8 + r) * stride + n0 + n] = f2bf(d[r]);
}

// ---------------------------------------------------------------------------
// Stage A1: fold the two linear layers: WkEff = Wk2@Wk1 (f32), WvEff = Wv2@Wv1
// (bf16), biases, bf16 copies of A/Bm, row-sums of A.
// ---------------------------------------------------------------------------
__global__ void __launch_bounds__(256) fuse1_kernel(
    const float* __restrict__ Wk1, const float* __restrict__ Wk2,
    const float* __restrict__ bk1, const float* __restrict__ bk2,
    const float* __restrict__ Wv1, const float* __restrict__ Wv2,
    const float* __restrict__ bv1, const float* __restrict__ bv2,
    const float* __restrict__ A,   const float* __restrict__ Bm,
    float* __restrict__ WKEFF, unsigned short* __restrict__ WVB,
    float* __restrict__ BKP, float* __restrict__ BV,
    unsigned short* __restrict__ ABF, unsigned short* __restrict__ BMBF,
    float* __restrict__ RSA) {
  int wg = blockIdx.x, t = threadIdx.x;
  int row = wg * 2 + (t >> 7);
  int col = t & 127;
  float accK = 0.f, accV = 0.f;
  for (int h = 0; h < 256; ++h) {
    accK += Wk2[row * 256 + h] * Wk1[h * 128 + col];
    accV += Wv2[row * 256 + h] * Wv1[h * 128 + col];
  }
  WKEFF[row * 128 + col] = accK;
  WVB[row * 128 + col] = f2bf(accV);
  int idx = wg * 256 + t;
  ABF[idx] = f2bf(A[idx]);
  BMBF[idx] = f2bf(Bm[idx]);
  if (wg == 0) {
    if (t < 128) {
      float a = 0.f;
      for (int h = 0; h < 256; ++h) a += Wk2[t * 256 + h] * bk1[h];
      BKP[t] = a + bk2[t];
    } else {
      int c = t - 128;
      float a = 0.f;
      for (int h = 0; h < 256; ++h) a += Wv2[c * 256 + h] * bv1[h];
      BV[c] = a + bv2[c];
    }
  } else if (wg == 1 && t < 128) {
    float a = 0.f;
    for (int v = 0; v < 128; ++v) a += A[t * 128 + v];
    RSA[t] = a;
  }
}

// ---------------------------------------------------------------------------
// Stage A2: M1 = WkEff[:, :64] @ Ws  (bf16), M2 = WkEff[:, 64:]  (bf16),
// BK = BKP + WkEff[:, :64] @ bs.
// ---------------------------------------------------------------------------
__global__ void __launch_bounds__(256) fuse2_kernel(
    const float* __restrict__ WKEFF, const float* __restrict__ Ws,
    const float* __restrict__ bs,    const float* __restrict__ BKP,
    unsigned short* __restrict__ M1, unsigned short* __restrict__ M2,
    float* __restrict__ BK) {
  int wg = blockIdx.x, t = threadIdx.x;
#pragma unroll
  for (int rep = 0; rep < 2; ++rep) {
    int idx = wg * 512 + rep * 256 + t;
    int row = idx >> 8, col = idx & 255;
    float a = 0.f;
    for (int m = 0; m < 64; ++m) a += WKEFF[row * 128 + m] * Ws[m * 256 + col];
    M1[idx] = f2bf(a);
  }
  if (t < 128) {
    int idx = wg * 128 + t;
    int row = idx >> 6, col = idx & 63;
    M2[idx] = f2bf(WKEFF[row * 128 + 64 + col]);
  }
  if (wg == 0 && t < 128) {
    float a = BKP[t];
    for (int m = 0; m < 64; ++m) a += WKEFF[t * 128 + m] * bs[m];
    BK[t] = a;
  }
}

// ---------------------------------------------------------------------------
// Stage B: activation GEMMs via WMMA.  128 rows per WG (8 waves x 16-row
// stripes).  k = relu(state@M1^T + tl@M2^T + bk), v = relu(value@WvEff^T+bv)*mod.
// Emits bf16 k/v and per-WG stat partial sums.
// ---------------------------------------------------------------------------
#define ACT_SX   0                       // 128*256 bf16 = 65536
#define ACT_ST   65536                   // 128*64  bf16 = 16384
#define ACT_SV   81920                   // 128*128 bf16 = 32768
#define ACT_MOD  114688                  // 128 f32      = 512
#define ACT_STAT 115200                  // 512 f32      = 2048
#define ACT_SMEM 117248

__global__ void __launch_bounds__(256) act_kernel(
    const float* __restrict__ state, const float* __restrict__ tl,
    const float* __restrict__ value, const float* __restrict__ mod,
    const unsigned short* __restrict__ gM1, const unsigned short* __restrict__ gM2,
    const unsigned short* __restrict__ gWv,
    const float* __restrict__ gbk, const float* __restrict__ gbv,
    unsigned short* __restrict__ gK, unsigned short* __restrict__ gV,
    float* __restrict__ gPart) {
  extern __shared__ char smem[];
  unsigned short* sX   = (unsigned short*)(smem + ACT_SX);
  unsigned short* sT   = (unsigned short*)(smem + ACT_ST);
  unsigned short* sVal = (unsigned short*)(smem + ACT_SV);
  float* sMod  = (float*)(smem + ACT_MOD);
  float* sStat = (float*)(smem + ACT_STAT);

  const int tid = threadIdx.x;
  const int lane = tid & 31;
  const int wv = tid >> 5;
  const int r0 = blockIdx.x * 128;

  for (int i = tid; i < 128 * 256; i += 256) sX[i] = f2bf(state[(size_t)r0 * 256 + i]);
  for (int i = tid; i < 128 * 64; i += 256)  sT[i] = f2bf(tl[(size_t)r0 * 64 + i]);
  for (int i = tid; i < 128 * 128; i += 256) sVal[i] = f2bf(value[(size_t)r0 * 128 + i]);
  if (tid < 128) sMod[tid] = mod[r0 + tid];
  for (int i = tid; i < 512; i += 256) sStat[i] = 0.f;
  __syncthreads();

  const int g = lane >> 4, n = lane & 15;
#pragma unroll
  for (int tj = 0; tj < 8; ++tj) {
    const int N0 = tj * 16;
    // ---- k path ----
    v8f acc = {0.f, 0.f, 0.f, 0.f, 0.f, 0.f, 0.f, 0.f};
#pragma unroll
    for (int kk = 0; kk < 256; kk += 32)
      acc = wmma_bf16(load_a_frag(sX, 256, wv * 16, kk, lane),
                      load_b_frag(gM1, 256, N0, kk, lane), acc);
#pragma unroll
    for (int kk = 0; kk < 64; kk += 32)
      acc = wmma_bf16(load_a_frag(sT, 64, wv * 16, kk, lane),
                      load_b_frag(gM2, 64, N0, kk, lane), acc);
    {
      float bk = gbk[N0 + n];
      float sk = 0.f, sk2 = 0.f;
#pragma unroll
      for (int r = 0; r < 8; ++r) {
        float x = acc[r] + bk;
        x = x > 0.f ? x : 0.f;
        int row = wv * 16 + g * 8 + r;
        gK[(size_t)(r0 + row) * 128 + N0 + n] = f2bf(x);
        sk += x; sk2 += x * x;
      }
      atomicAdd(&sStat[N0 + n], sk);
      atomicAdd(&sStat[128 + N0 + n], sk2);
    }
    // ---- v path ----
    v8f av = {0.f, 0.f, 0.f, 0.f, 0.f, 0.f, 0.f, 0.f};
#pragma unroll
    for (int kk = 0; kk < 128; kk += 32)
      av = wmma_bf16(load_a_frag(sVal, 128, wv * 16, kk, lane),
                     load_b_frag(gWv, 128, N0, kk, lane), av);
    {
      float bv = gbv[N0 + n];
      float sv = 0.f, sv2 = 0.f;
#pragma unroll
      for (int r = 0; r < 8; ++r) {
        int row = wv * 16 + g * 8 + r;
        float x = av[r] + bv;
        x = x > 0.f ? x : 0.f;
        x *= sMod[row];
        gV[(size_t)(r0 + row) * 128 + N0 + n] = f2bf(x);
        sv += x; sv2 += x * x;
      }
      atomicAdd(&sStat[256 + N0 + n], sv);
      atomicAdd(&sStat[384 + N0 + n], sv2);
    }
  }
  __syncthreads();
  for (int i = tid; i < 512; i += 256) gPart[(size_t)blockIdx.x * 512 + i] = sStat[i];
}

// ---------------------------------------------------------------------------
// Stage C: reduce 1024 partials -> running mean / inv-std (Welford update from
// mean=0, var=1, count=1e-4).
// ---------------------------------------------------------------------------
__global__ void __launch_bounds__(128) stats_kernel(const float* __restrict__ gPart,
                                                    float* __restrict__ gStats) {
  int c = threadIdx.x;
  float sk = 0.f, sk2 = 0.f, sv = 0.f, sv2 = 0.f;
  for (int p = 0; p < 1024; ++p) {
    const float* pp = gPart + (size_t)p * 512;
    sk += pp[c]; sk2 += pp[128 + c];
    sv += pp[256 + c]; sv2 += pp[384 + c];
  }
  const float n = (float)NROWS, ic = 1e-4f, tot = n + ic;
  {
    float bm = sk / n;
    float bv = sk2 / n - bm * bm;
    float mean = bm * n / tot;
    float m2 = ic * 1.0f + bv * n + bm * bm * (ic * n / tot);
    gStats[c] = mean;
    gStats[128 + c] = rsqrtf(m2 / tot + 1e-8f);
  }
  {
    float bm = sv / n;
    float bv = sv2 / n - bm * bm;
    float mean = bm * n / tot;
    float m2 = ic * 1.0f + bv * n + bm * bm * (ic * n / tot);
    gStats[256 + c] = mean;
    gStats[384 + c] = rsqrtf(m2 / tot + 1e-8f);
  }
}

// ---------------------------------------------------------------------------
// Stage D: per-batch Hebbian iterations.  One WG (8 waves) per batch.
// w kept in registers (f32 WMMA accumulators), bf16 operands staged in LDS,
// A/Bm/k/v staged via gfx1250 async global->LDS (ASYNCcnt) when available.
//   corrT = vn^T@kn,  reg = kn^T@kn  (T=32 -> one WMMA per tile)
//   per iter: P = A@w^T -> a1 = rowsumA - P;  a3 = Bm@w^T
//             w += LR*(a1@corr);  a4 = (a3@reg) stored transposed; w -= LR*a4
// ---------------------------------------------------------------------------
#define HB_CORRT 0                       // 32768
#define HB_REG   32768                   // 32768
#define HB_SA    65536                   // 32768
#define HB_SBM   98304                   // 32768
#define HB_SWBF  131072                  // 32768 (aliases k/v raw staging)
#define HB_SA1   163840                  // 32768 (aliases KnT: 128x32 bf16)
#define HB_SA3   196608                  // 32768 (aliases VnT)
#define HB_SA4   229376                  // 65536 (f32)
#define HB_RS    294912                  // 512 (f32)
#define HB_SMEM  295424

__global__ void __launch_bounds__(256) hebb_kernel(
    const float* __restrict__ w_assoc,
    const unsigned short* __restrict__ gK, const unsigned short* __restrict__ gV,
    const float* __restrict__ gStats,
    const unsigned short* __restrict__ gAbf, const unsigned short* __restrict__ gBmbf,
    const float* __restrict__ gRowSumA,
    float* __restrict__ out) {
  extern __shared__ char smem[];
  unsigned short* sCorrT = (unsigned short*)(smem + HB_CORRT);
  unsigned short* sReg   = (unsigned short*)(smem + HB_REG);
  unsigned short* sA     = (unsigned short*)(smem + HB_SA);
  unsigned short* sBm    = (unsigned short*)(smem + HB_SBM);
  unsigned short* sWbf   = (unsigned short*)(smem + HB_SWBF);
  unsigned short* sA1    = (unsigned short*)(smem + HB_SA1);
  unsigned short* sA3    = (unsigned short*)(smem + HB_SA3);
  float* sA4 = (float*)(smem + HB_SA4);
  float* sRS = (float*)(smem + HB_RS);
  unsigned short* sKnT = sA1;        // alias: only live before the iterations
  unsigned short* sVnT = sA3;
  unsigned short* sStgK = sWbf;      // raw bf16 k rows (32x128), dead at iter (a)
  unsigned short* sStgV = sWbf + TT * KD;

  const int b = blockIdx.x;
  const int tid = threadIdx.x;
  const int lane = tid & 31;
  const int wv = tid >> 5;
  const int g = lane >> 4, n = lane & 15;

  // ---- issue async global->LDS staging (A, Bm, raw k/v rows) ----
  {
    const unsigned short* kb = gK + (size_t)b * TT * KD;
    const unsigned short* vb = gV + (size_t)b * TT * VD;
#pragma unroll
    for (int rep = 0; rep < 8; ++rep) {
      int i = rep * 256 + tid;                 // b128 transfer index
      async_b128(sA + i * 8, gAbf + i * 8);
      async_b128(sBm + i * 8, gBmbf + i * 8);
    }
#pragma unroll
    for (int rep = 0; rep < 2; ++rep) {
      int i = rep * 256 + tid;
      async_b128(sStgK + i * 8, kb + i * 8);
      async_b128(sStgV + i * 8, vb + i * 8);
    }
  }

  // ---- load w into register fragments (overlaps async staging) ----
  v8f wfrag[8];
  {
    const float* wb = w_assoc + (size_t)b * 16384;
#pragma unroll
    for (int tj = 0; tj < 8; ++tj) {
      v8f f;
#pragma unroll
      for (int r = 0; r < 8; ++r)
        f[r] = wb[(wv * 16 + g * 8 + r) * 128 + tj * 16 + n];
      wfrag[tj] = f;
    }
  }
  if (tid < 128) sRS[tid] = gRowSumA[tid];

  async_wait0();
  __syncthreads();

  // ---- normalize kn/vn into transposed LDS layout [dim][t] ----
  const float* kmean = gStats;       const float* kistd = gStats + 128;
  const float* vmean = gStats + 256; const float* vistd = gStats + 384;
  for (int i = tid; i < TT * KD; i += 256) {
    int c = i & 127, t = i >> 7;
    sKnT[c * TT + t] = f2bf((bf2f(sStgK[i]) - kmean[c]) * kistd[c]);
    sVnT[c * TT + t] = f2bf((bf2f(sStgV[i]) - vmean[c]) * vistd[c]);
  }
  __syncthreads();

  // ---- corrT = vn^T @ kn, reg = kn^T @ kn (single K=32 WMMA per tile) ----
#pragma unroll
  for (int tj = 0; tj < 8; ++tj) {
    v8f z = {0.f, 0.f, 0.f, 0.f, 0.f, 0.f, 0.f, 0.f};
    v16bf bb = load_b_frag(sKnT, TT, tj * 16, 0, lane);
    v8f d = wmma_bf16(load_a_frag(sVnT, TT, wv * 16, 0, lane), bb, z);
    store_tile_bf(sCorrT, 128, wv * 16, tj * 16, d, lane);
    v8f d2 = wmma_bf16(load_a_frag(sKnT, TT, wv * 16, 0, lane), bb, z);
    store_tile_bf(sReg, 128, wv * 16, tj * 16, d2, lane);
  }
  __syncthreads();

  // ---- 4 Hebbian iterations ----
  for (int it = 0; it < 4; ++it) {
    // (a) refresh bf16 w in LDS from register fragments
#pragma unroll
    for (int tj = 0; tj < 8; ++tj)
#pragma unroll
      for (int r = 0; r < 8; ++r)
        sWbf[(wv * 16 + g * 8 + r) * 128 + tj * 16 + n] = f2bf(wfrag[tj][r]);
    __syncthreads();

    // (b) a1 = rowsumA - A@w^T ; a3 = Bm@w^T
#pragma unroll
    for (int tj = 0; tj < 8; ++tj) {
      v8f acc  = {0.f, 0.f, 0.f, 0.f, 0.f, 0.f, 0.f, 0.f};
      v8f acc3 = {0.f, 0.f, 0.f, 0.f, 0.f, 0.f, 0.f, 0.f};
#pragma unroll
      for (int kk = 0; kk < 128; kk += 32) {
        v16bf bfr = load_b_frag(sWbf, 128, tj * 16, kk, lane);
        acc  = wmma_bf16(load_a_frag(sA, 128, wv * 16, kk, lane), bfr, acc);
        acc3 = wmma_bf16(load_a_frag(sBm, 128, wv * 16, kk, lane), bfr, acc3);
      }
#pragma unroll
      for (int r = 0; r < 8; ++r) {
        int row = wv * 16 + g * 8 + r;
        sA1[row * 128 + tj * 16 + n] = f2bf(sRS[row] - acc[r]);
        sA3[row * 128 + tj * 16 + n] = f2bf(acc3[r]);
      }
    }
    __syncthreads();

    // (c) w += LR*(a1@corr) ; stage (a3@reg) transposed into sA4
#pragma unroll
    for (int tj = 0; tj < 8; ++tj) {
      v8f acc  = {0.f, 0.f, 0.f, 0.f, 0.f, 0.f, 0.f, 0.f};
      v8f acc4 = {0.f, 0.f, 0.f, 0.f, 0.f, 0.f, 0.f, 0.f};
#pragma unroll
      for (int kk = 0; kk < 128; kk += 32) {
        acc  = wmma_bf16(load_a_frag(sA1, 128, wv * 16, kk, lane),
                         load_b_frag(sCorrT, 128, tj * 16, kk, lane), acc);
        acc4 = wmma_bf16(load_a_frag(sA3, 128, wv * 16, kk, lane),
                         load_b_frag(sReg, 128, tj * 16, kk, lane), acc4);
      }
#pragma unroll
      for (int r = 0; r < 8; ++r) {
        wfrag[tj][r] += LRATE * acc[r];
        // element (u = wv*16+g*8+r, j = tj*16+n) of a3@reg -> sA4[j][u]
        sA4[(tj * 16 + n) * 128 + (wv * 16 + g * 8 + r)] = acc4[r];
      }
    }
    __syncthreads();

    // (d) w -= LR * a4   (sA4 now indexed [k][v])
#pragma unroll
    for (int tj = 0; tj < 8; ++tj)
#pragma unroll
      for (int r = 0; r < 8; ++r)
        wfrag[tj][r] -= LRATE * sA4[(wv * 16 + g * 8 + r) * 128 + tj * 16 + n];
    __syncthreads();
  }

  // ---- write result ----
  float* ob = out + (size_t)b * 16384;
#pragma unroll
  for (int tj = 0; tj < 8; ++tj)
#pragma unroll
    for (int r = 0; r < 8; ++r)
      ob[(wv * 16 + g * 8 + r) * 128 + tj * 16 + n] = wfrag[tj][r];
}

// ---------------------------------------------------------------------------
// Host-side launch
// ---------------------------------------------------------------------------
extern "C" void kernel_launch(void* const* d_in, const int* in_sizes, int n_in,
                              void* d_out, int out_size, void* d_ws, size_t ws_size,
                              hipStream_t stream) {
  (void)in_sizes; (void)n_in; (void)out_size; (void)ws_size;
  const float* state = (const float*)d_in[0];
  const float* tl    = (const float*)d_in[1];
  const float* value = (const float*)d_in[2];
  const float* mod   = (const float*)d_in[3];
  const float* w0    = (const float*)d_in[4];
  const float* A     = (const float*)d_in[5];
  const float* Bm    = (const float*)d_in[6];
  const float* Ws    = (const float*)d_in[7];
  const float* bs    = (const float*)d_in[8];
  const float* Wk1   = (const float*)d_in[9];
  const float* bk1   = (const float*)d_in[10];
  const float* Wk2   = (const float*)d_in[11];
  const float* bk2   = (const float*)d_in[12];
  const float* Wv1   = (const float*)d_in[13];
  const float* bv1   = (const float*)d_in[14];
  const float* Wv2   = (const float*)d_in[15];
  const float* bv2   = (const float*)d_in[16];

  char* ws = (char*)d_ws;
  size_t o = 0;
  auto alloc = [&](size_t bytes) { char* p = ws + o; o = (o + bytes + 255) & ~(size_t)255; return p; };
  float*          WKEFF = (float*)alloc(16384 * 4);
  unsigned short* M1    = (unsigned short*)alloc(32768 * 2);
  unsigned short* M2    = (unsigned short*)alloc(8192 * 2);
  unsigned short* WVB   = (unsigned short*)alloc(16384 * 2);
  float*          BKP   = (float*)alloc(128 * 4);
  float*          BK    = (float*)alloc(128 * 4);
  float*          BV    = (float*)alloc(128 * 4);
  unsigned short* ABF   = (unsigned short*)alloc(16384 * 2);
  unsigned short* BMBF  = (unsigned short*)alloc(16384 * 2);
  float*          RSA   = (float*)alloc(128 * 4);
  float*          STATS = (float*)alloc(512 * 4);
  float*          PART  = (float*)alloc((size_t)1024 * 512 * 4);
  unsigned short* GK    = (unsigned short*)alloc((size_t)NROWS * KD * 2);
  unsigned short* GV    = (unsigned short*)alloc((size_t)NROWS * VD * 2);

  fuse1_kernel<<<64, 256, 0, stream>>>(Wk1, Wk2, bk1, bk2, Wv1, Wv2, bv1, bv2,
                                       A, Bm, WKEFF, WVB, BKP, BV, ABF, BMBF, RSA);
  fuse2_kernel<<<64, 256, 0, stream>>>(WKEFF, Ws, bs, BKP, M1, M2, BK);
  act_kernel<<<1024, 256, ACT_SMEM, stream>>>(state, tl, value, mod, M1, M2, WVB,
                                              BK, BV, GK, GV, PART);
  stats_kernel<<<1, 128, 0, stream>>>(PART, STATS);
  hebb_kernel<<<BSZ, 256, HB_SMEM, stream>>>(w0, GK, GV, STATS, ABF, BMBF, RSA,
                                             (float*)d_out);
}
